// MultiHeadAttention_54906861912476
// MI455X (gfx1250) — matrix operand order repaired and checked by reading, not compiled, not run on hardware
//
#include <hip/hip_runtime.h>

// Problem constants (match reference)
#define B_   4
#define N_   2048
#define DIM_ 1024
#define H_   16
#define D_   64
#define F_   3072   // 3*H*D

typedef __attribute__((ext_vector_type(16))) __bf16 v16bf;
typedef __attribute__((ext_vector_type(8)))  float  v8f;

union ABReg { unsigned int u[8]; v16bf v; };

__device__ __forceinline__ unsigned short bf16_1(float x) {
  unsigned int u = __float_as_uint(x);
  u = u + 0x7FFFu + ((u >> 16) & 1u);        // round-to-nearest-even
  return (unsigned short)(u >> 16);
}

__device__ __forceinline__ unsigned int pack2bf(float lo, float hi) {
  unsigned int a = __float_as_uint(lo);
  unsigned int b = __float_as_uint(hi);
  a = (a + 0x7FFFu + ((a >> 16) & 1u)) >> 16;
  b = (b + 0x7FFFu + ((b >> 16) & 1u)) & 0xFFFF0000u;
  return (a & 0xFFFFu) | b;
}

__device__ __forceinline__ v8f zero_v8f() {
  v8f z;
  #pragma unroll
  for (int i = 0; i < 8; ++i) z[i] = 0.0f;
  return z;
}

// ---------------------------------------------------------------------------
// Kernel 0: bulk fp32 -> bf16 conversion (bandwidth-bound, runs once).
// Each thread converts 8 floats -> 4 packed uints (b128 in, b128 out).
// ---------------------------------------------------------------------------
__global__ __launch_bounds__(256) void f32_to_bf16_kernel(
    const float* __restrict__ src, unsigned int* __restrict__ dst, int n8)
{
  const int i = blockIdx.x * 256 + threadIdx.x;
  if (i >= n8) return;
  const float4* s = (const float4*)src + 2 * (size_t)i;
  const float4 a = s[0];
  const float4 b = s[1];
  uint4 o;
  o.x = pack2bf(a.x, a.y);
  o.y = pack2bf(a.z, a.w);
  o.z = pack2bf(b.x, b.y);
  o.w = pack2bf(b.z, b.w);
  ((uint4*)dst)[i] = o;
}

// ---------------------------------------------------------------------------
// Kernel 1: qkv = x @ Wqkv^T   (M=B*N=8192, F=3072, K=1024), bf16 WMMA.
// Each wave computes a 64(M) x 64(F) tile: 16 WMMAs per 32-wide k-step,
// operands loaded as two uint4 (global_load_b128) per 16x32 tile. Outputs:
//   q  [B,H,N,D] bf16 (pre-scaled by 1/sqrt(D)=0.125)
//   k  [B,H,N,D] bf16
//   vt [B,H,D,N] bf16 (transposed so PV B-operand is contiguous along keys)
// ---------------------------------------------------------------------------
__global__ __launch_bounds__(256) void qkv_gemm_kernel(
    const unsigned int* __restrict__ xb, const unsigned int* __restrict__ Wb,
    unsigned short* __restrict__ q, unsigned short* __restrict__ kmat,
    unsigned short* __restrict__ vt)
{
  const int lane = threadIdx.x & 31;
  const int wave = threadIdx.x >> 5;
  const int task = blockIdx.x * 8 + wave;      // 6144 wave-tasks total
  const int FT   = F_ / 64;                    // 48
  const int ft   = task % FT;
  const int mt   = task / FT;                  // < 128
  const int m0   = mt * 64;
  const int f0   = ft * 64;
  const int lr   = lane & 15;
  const int hi   = lane >> 4;
  const int KW   = DIM_ / 2;                   // 512 uints (bf16 pairs) per row

  v8f acc[4][4];
  #pragma unroll
  for (int t = 0; t < 4; ++t)
    #pragma unroll
    for (int c = 0; c < 4; ++c) acc[t][c] = zero_v8f();

  // Row base pointers already include the lane-group K offsets:
  // A-layout uint index = kk/2 + r + 8*hi
  // B-layout uint index = kk/2 + r + 8*hi
  const unsigned int* xr[4];
  #pragma unroll
  for (int t = 0; t < 4; ++t)
    xr[t] = xb + (size_t)(m0 + 16 * t + lr) * KW + 8 * hi;
  const unsigned int* wr[4];
  #pragma unroll
  for (int c = 0; c < 4; ++c)
    wr[c] = Wb + (size_t)(f0 + 16 * c + lr) * KW + 8 * hi;

  for (int kw = 0; kw < KW; kw += 16) {        // 16 uints = 32 K values
    // Prefetch next k-tile (global_prefetch_b8) to cover L2 latency.
    if (kw + 16 < KW) {
      #pragma unroll
      for (int c = 0; c < 4; ++c) __builtin_prefetch(wr[c] + kw + 16, 0, 1);
      #pragma unroll
      for (int t = 0; t < 4; ++t) __builtin_prefetch(xr[t] + kw + 16, 0, 1);
    }

    ABReg a[4];
    #pragma unroll
    for (int t = 0; t < 4; ++t) {
      const uint4 lo = *(const uint4*)(xr[t] + kw);       // r=0..3
      const uint4 hh = *(const uint4*)(xr[t] + kw + 4);   // r=4..7
      a[t].u[0] = lo.x; a[t].u[1] = lo.y; a[t].u[2] = lo.z; a[t].u[3] = lo.w;
      a[t].u[4] = hh.x; a[t].u[5] = hh.y; a[t].u[6] = hh.z; a[t].u[7] = hh.w;
    }
    #pragma unroll
    for (int c = 0; c < 4; ++c) {
      ABReg bb;
      const uint4 lo = *(const uint4*)(wr[c] + kw);
      const uint4 hh = *(const uint4*)(wr[c] + kw + 4);
      bb.u[0] = lo.x; bb.u[1] = lo.y; bb.u[2] = lo.z; bb.u[3] = lo.w;
      bb.u[4] = hh.x; bb.u[5] = hh.y; bb.u[6] = hh.z; bb.u[7] = hh.w;
      #pragma unroll
      for (int t = 0; t < 4; ++t)
        acc[t][c] = __builtin_amdgcn_wmma_f32_16x16x32_bf16(
            false, a[t].v, false, bb.v, (short)0, acc[t][c], false, false);
    }
  }

  // Scatter results to q / k / v^T in bf16. C-layout: VGPR j -> row j+8*hi, col lr.
  #pragma unroll
  for (int c = 0; c < 4; ++c) {
    const int fbase = f0 + 16 * c;
    const int sel   = fbase >> 10;          // 0=q, 1=k, 2=v
    const int h     = (fbase & 1023) >> 6;
    const int d     = (fbase & 63) + lr;
    #pragma unroll
    for (int t = 0; t < 4; ++t) {
      #pragma unroll
      for (int j = 0; j < 8; ++j) {
        const int m  = m0 + 16 * t + j + 8 * hi;
        const int bb = m >> 11;             // m / N_
        const int n  = m & (N_ - 1);
        const float val = acc[t][c][j];
        const size_t qoff = ((size_t)(bb * H_ + h) * N_ + n) * D_ + d;
        if (sel == 0)      q[qoff]    = bf16_1(val * 0.125f);   // fold 1/sqrt(D)
        else if (sel == 1) kmat[qoff] = bf16_1(val);
        else vt[((size_t)(bb * H_ + h) * D_ + d) * N_ + n] = bf16_1(val);
      }
    }
  }
}

// ---------------------------------------------------------------------------
// Kernel 2: flash attention. One wave per (b, h, 16-row tile). 32-key blocks:
// S = Q K^T via 4 WMMAs, masks + online softmax in C-layout, P re-layout via
// per-wave LDS, O += P V via 4 WMMAs against v^T. Epilogue divides by l
// (l==0 -> 0 matches nan_to_num for fully masked rows).
// ---------------------------------------------------------------------------
__global__ __launch_bounds__(256) void attn_kernel(
    const unsigned short* __restrict__ q, const unsigned short* __restrict__ kmat,
    const unsigned short* __restrict__ vt, const int* __restrict__ key_mask,
    const int* __restrict__ query_mask, const int* __restrict__ causal_p,
    float* __restrict__ out)
{
  __shared__ unsigned short Pbuf[8 * 512];   // 1KB per wave: 16x32 bf16 P tile

  const int lane = threadIdx.x & 31;
  const int wave = threadIdx.x >> 5;
  const int task = blockIdx.x * 8 + wave;    // 8192 tasks
  const int rt   = task & 127;               // N/16 = 128 row tiles
  const int bh   = task >> 7;
  const int h    = bh & (H_ - 1);
  const int b    = bh >> 4;
  const int r0   = rt * 16;
  const int lr   = lane & 15;
  const int hi   = lane >> 4;
  const int causal = causal_p[0];

  const size_t qk_base = (size_t)(b * H_ + h) * N_ * D_;
  const size_t v_base  = (size_t)(b * H_ + h) * D_ * N_;
  unsigned short* pb   = &Pbuf[wave * 512];

  // Q tile in A-layout, both K halves (D=64 -> two 32-wide K steps)
  ABReg qa[2];
  #pragma unroll
  for (int s = 0; s < 2; ++s) {
    #pragma unroll
    for (int r = 0; r < 8; ++r) {
      const int kidx = 32 * s + 2 * r + 16 * hi;
      qa[s].u[r] = *(const unsigned int*)(q + qk_base + (size_t)(r0 + lr) * D_ + kidx);
    }
  }

  int qm[8];
  #pragma unroll
  for (int j = 0; j < 8; ++j) qm[j] = query_mask[b * N_ + r0 + j + 8 * hi];

  const float NEGINF = -__builtin_inff();
  float mrow[8], lrow[8];
  #pragma unroll
  for (int j = 0; j < 8; ++j) { mrow[j] = NEGINF; lrow[j] = 0.0f; }
  v8f oacc[4];
  #pragma unroll
  for (int c = 0; c < 4; ++c) oacc[c] = zero_v8f();

  const int jend = causal ? (r0 + 16) : N_;  // causal: keys only up to diagonal

  for (int j0 = 0; j0 < jend; j0 += 32) {
    // ---- S = Q K^T (two 16-col tiles) ----
    v8f s0 = zero_v8f(), s1 = zero_v8f();
    #pragma unroll
    for (int c = 0; c < 2; ++c) {
      v8f sc = zero_v8f();
      #pragma unroll
      for (int s = 0; s < 2; ++s) {
        ABReg kb;
        const unsigned short* kr =
            kmat + qk_base + (size_t)(j0 + 16 * c + lr) * D_ + 32 * s + 16 * hi;
        #pragma unroll
        for (int r = 0; r < 8; ++r) kb.u[r] = *(const unsigned int*)(kr + 2 * r);
        sc = __builtin_amdgcn_wmma_f32_16x16x32_bf16(
            false, qa[s].v, false, kb.v, (short)0, sc, false, false);
      }
      if (c == 0) s0 = sc; else s1 = sc;
    }

    // ---- masks + online softmax (C-layout: row = j+8*hi, col = lane&15) ----
    const int km0 = key_mask[b * N_ + j0 + lr];
    const int km1 = key_mask[b * N_ + j0 + 16 + lr];
    float alpha[8], p0[8], p1[8];
    #pragma unroll
    for (int j = 0; j < 8; ++j) {
      const int row = r0 + j + 8 * hi;
      const int c0  = j0 + lr;
      const int c1  = j0 + 16 + lr;
      float a0 = s0[j], a1 = s1[j];
      const bool ok0 = km0 && qm[j] && (!causal || c0 <= row);
      const bool ok1 = km1 && qm[j] && (!causal || c1 <= row);
      a0 = ok0 ? a0 : NEGINF;
      a1 = ok1 ? a1 : NEGINF;

      float tm = fmaxf(a0, a1);
      #pragma unroll
      for (int msk = 1; msk < 16; msk <<= 1)
        tm = fmaxf(tm, __shfl_xor(tm, msk, 32));   // reduce within 16-lane group

      const float mn    = fmaxf(mrow[j], tm);
      const float msafe = (mn == NEGINF) ? 0.0f : mn;
      const float al    = (mrow[j] == NEGINF) ? 0.0f : __expf(mrow[j] - mn);
      const float e0 = __expf(a0 - msafe);         // -inf -> 0
      const float e1 = __expf(a1 - msafe);

      float ps = e0 + e1;
      #pragma unroll
      for (int msk = 1; msk < 16; msk <<= 1)
        ps += __shfl_xor(ps, msk, 32);

      lrow[j] = lrow[j] * al + ps;
      mrow[j] = mn;
      alpha[j] = al;
      p0[j] = e0; p1[j] = e1;
    }
    #pragma unroll
    for (int c2 = 0; c2 < 4; ++c2)
      #pragma unroll
      for (int j = 0; j < 8; ++j) oacc[c2][j] *= alpha[j];

    // ---- re-layout P: C-layout -> LDS -> A-layout (same-wave DS is in-order) ----
    #pragma unroll
    for (int j = 0; j < 8; ++j) {
      const int row = j + 8 * hi;
      pb[row * 32 + lr]      = bf16_1(p0[j]);
      pb[row * 32 + 16 + lr] = bf16_1(p1[j]);
    }
    ABReg pa;
    #pragma unroll
    for (int r = 0; r < 8; ++r) {
      const int kidx = 2 * r + 16 * hi;
      const unsigned int lo = pb[lr * 32 + kidx];
      const unsigned int hw = pb[lr * 32 + kidx + 1];
      pa.u[r] = lo | (hw << 16);
    }

    // ---- O += P V  (B operand from v^T: contiguous along key index) ----
    #pragma unroll
    for (int c2 = 0; c2 < 4; ++c2) {
      ABReg vb;
      const unsigned short* vr =
          vt + v_base + (size_t)(16 * c2 + lr) * N_ + j0 + 16 * hi;
      #pragma unroll
      for (int r = 0; r < 8; ++r) vb.u[r] = *(const unsigned int*)(vr + 2 * r);
      oacc[c2] = __builtin_amdgcn_wmma_f32_16x16x32_bf16(
          false, pa.v, false, vb.v, (short)0, oacc[c2], false, false);
    }
  }

  // ---- epilogue: out[b][n][h*D + d] = O / l  (l==0 -> 0) ----
  #pragma unroll
  for (int c2 = 0; c2 < 4; ++c2) {
    #pragma unroll
    for (int j = 0; j < 8; ++j) {
      const int row = r0 + j + 8 * hi;
      const float l = lrow[j];
      const float val = (l > 0.0f) ? (oacc[c2][j] / l) : 0.0f;
      out[(size_t)(b * N_ + row) * (H_ * D_) + h * D_ + 16 * c2 + lr] = val;
    }
  }
}

extern "C" void kernel_launch(void* const* d_in, const int* in_sizes, int n_in,
                              void* d_out, int out_size, void* d_ws, size_t ws_size,
                              hipStream_t stream) {
  (void)in_sizes; (void)n_in; (void)out_size; (void)ws_size;
  const float* x          = (const float*)d_in[0];
  const float* W          = (const float*)d_in[1];
  const int*   key_mask   = (const int*)d_in[2];
  const int*   query_mask = (const int*)d_in[3];
  const int*   causal     = (const int*)d_in[4];

  // Workspace layout (bytes):
  //   xb : B*N*DIM bf16        = 16 MB
  //   Wb : F*DIM bf16          =  6 MB
  //   q  : B*H*N*D bf16        = 16 MB
  //   k  : B*H*N*D bf16        = 16 MB
  //   vt : B*H*D*N bf16        = 16 MB   (total 70 MB)
  unsigned int* xb = (unsigned int*)d_ws;                       // bf16 pairs
  unsigned int* Wb = xb + (size_t)B_ * N_ * DIM_ / 2;
  unsigned short* q  = (unsigned short*)(Wb + (size_t)F_ * DIM_ / 2);
  unsigned short* k  = q + (size_t)B_ * H_ * N_ * D_;
  unsigned short* vt = k + (size_t)B_ * H_ * N_ * D_;

  // fp32 -> bf16 bulk conversion (8 floats per thread)
  const int n8x = B_ * N_ * DIM_ / 8;     // 1048576
  const int n8w = F_ * DIM_ / 8;          // 393216
  f32_to_bf16_kernel<<<(n8x + 255) / 256, 256, 0, stream>>>(x, xb, n8x);
  f32_to_bf16_kernel<<<(n8w + 255) / 256, 256, 0, stream>>>(W, Wb, n8w);

  // GEMM: (8192/64) * (3072/64) = 6144 wave-tasks / 8 waves per block
  qkv_gemm_kernel<<<768, 256, 0, stream>>>(xb, Wb, q, k, vt);
  // Attention: B*H*(N/16) = 8192 wave-tasks / 8 waves per block
  attn_kernel<<<1024, 256, 0, stream>>>(q, k, vt, key_mask, query_mask, causal,
                                        (float*)d_out);
}
